// DeformableTransformerDecoderLayer_one2one_57346403336779
// MI455X (gfx1250) — matrix operand order repaired and checked by reading, not compiled, not run on hardware
//
#include <hip/hip_runtime.h>
#include <hip/hip_bf16.h>
#include <math.h>

// ---------------------------------------------------------------------------
// Deformable Transformer decoder layer (one2one) for gfx1250 / MI455X.
// Dense GEMMs AND both self-attentions run on v_wmma_f32_16x16x32_f16.
// ---------------------------------------------------------------------------

#define D_MODEL 256
#define D_FFN   2048
#define NHEADS  8
#define BATCH   4
#define LQ      900
#define LV      17821
#define MQ      (LQ * BATCH)       // 3600 query rows
#define MV      (LV * BATCH)       // 71284 memory rows
#define SP      928                // padded key stride (29*32) for score rows

typedef __attribute__((ext_vector_type(16))) _Float16 v16h;
typedef __attribute__((ext_vector_type(8)))  _Float16 v8h;
typedef __attribute__((ext_vector_type(4)))  _Float16 v4h;
typedef __attribute__((ext_vector_type(8)))  float    v8f;

// ---------------------------------------------------------------------------
// WMMA GEMM:  C[M,N] = act(A[M,K] @ W[N,K]^T + bias[N])
// block = 256 threads = 8 waves; block tile 64(M) x 64(N); K step 32.
// Each wave computes 16x32 (2 WMMAs sharing one A fragment).
// Requires: N % 64 == 0, K % 32 == 0. M ragged OK. Double-buffered LDS.
// ---------------------------------------------------------------------------
__global__ void gemm_wmma_kernel(const float* __restrict__ A,
                                 const float* __restrict__ W,
                                 const float* __restrict__ bias,
                                 float* __restrict__ Cout,
                                 int M, int N, int K, int relu) {
  __shared__ alignas(64) _Float16 sA[2][64 * 32];   // [m][k]
  __shared__ alignas(64) _Float16 sB[2][64 * 32];   // [n][k]

  const int tid   = threadIdx.x;
  const int lane  = tid & 31;
  const int wave  = tid >> 5;
  const int hi    = lane >> 4;
  const int ln16  = lane & 15;
  const int mBlk  = blockIdx.x * 64;
  const int nBlk  = blockIdx.y * 64;
  const int mW    = (wave & 3) * 16;
  const int nW    = (wave >> 2) * 32;

  float4 ra[2], rb[2];
  const int r0  = tid >> 3;             // 0..31
  const int cc4 = (tid & 7) * 4;        // 0,4,..,28

  auto load_tile = [&](int k0) {
#pragma unroll
    for (int i = 0; i < 2; ++i) {
      int r = r0 + i * 32;
      int m = mBlk + r;
      if (m < M) ra[i] = *(const float4*)(A + (size_t)m * K + k0 + cc4);
      else       ra[i] = make_float4(0.f, 0.f, 0.f, 0.f);
      int n = nBlk + r;
      rb[i] = *(const float4*)(W + (size_t)n * K + k0 + cc4);
    }
  };
  auto store_tile = [&](int buf) {
#pragma unroll
    for (int i = 0; i < 2; ++i) {
      int r = r0 + i * 32;
      v4h pa; pa[0] = (_Float16)ra[i].x; pa[1] = (_Float16)ra[i].y;
              pa[2] = (_Float16)ra[i].z; pa[3] = (_Float16)ra[i].w;
      *(v4h*)&sA[buf][r * 32 + cc4] = pa;
      v4h pb; pb[0] = (_Float16)rb[i].x; pb[1] = (_Float16)rb[i].y;
              pb[2] = (_Float16)rb[i].z; pb[3] = (_Float16)rb[i].w;
      *(v4h*)&sB[buf][r * 32 + cc4] = pb;
    }
  };

  v8f cacc[2] = {{}, {}};
  const int nk = K >> 5;
  int buf = 0;
  load_tile(0);

  for (int kt = 0; kt < nk; ++kt) {
    store_tile(buf);
    __syncthreads();
    if (kt + 1 < nk) load_tile((kt + 1) << 5);

    const v8h* arow = (const v8h*)&sA[buf][(mW + ln16) * 32];
    v8h alo = arow[2 * hi];
    v8h ahi = arow[2 * hi + 1];
    v16h a;
#pragma unroll
    for (int i = 0; i < 8; ++i) { a[i] = alo[i]; a[8 + i] = ahi[i]; }

#pragma unroll
    for (int j = 0; j < 2; ++j) {
      v16h b = *(const v16h*)&sB[buf][(nW + j * 16 + ln16) * 32 + hi * 16];
      cacc[j] = __builtin_amdgcn_wmma_f32_16x16x32_f16(false, a, false, b,
                                                       (short)0, cacc[j],
                                                       false, false);
    }
    buf ^= 1;
  }

#pragma unroll
  for (int j = 0; j < 2; ++j) {
    int n = nBlk + nW + j * 16 + ln16;
    float bs = bias[n];
#pragma unroll
    for (int r = 0; r < 8; ++r) {
      int m = mBlk + mW + r + hi * 8;
      if (m < M) {
        float v = cacc[j][r] + bs;
        if (relu) v = fmaxf(v, 0.f);
        Cout[(size_t)m * N + n] = v;
      }
    }
  }
}

// ---------------------------------------------------------------------------
// Attention stage 1: S[bh][m][n] = scale * Q[m]·K[n]   (f16 output, SP stride)
// blockIdx.z = bh; tile 64(queries) x 64(keys); dh=32 => single WMMA K-step.
// qkv layout [LQ,B,768]: row (m*BATCH+b), q|k|v at +0/+256/+512, head +h*32.
// ---------------------------------------------------------------------------
__global__ void attn_qk_kernel(const float* __restrict__ qkv,
                               _Float16* __restrict__ S) {
  __shared__ alignas(64) _Float16 sA[64 * 32];
  __shared__ alignas(64) _Float16 sB[64 * 32];

  const int z = blockIdx.z;
  const int b = z >> 3, h = z & 7;
  const float* Qb = qkv + (size_t)b * 768 + h * 32;
  const float* Kb = Qb + 256;

  const int tid  = threadIdx.x;
  const int lane = tid & 31;
  const int wave = tid >> 5;
  const int hi   = lane >> 4;
  const int ln16 = lane & 15;
  const int mBlk = blockIdx.x * 64;
  const int nBlk = blockIdx.y * 64;
  const int mW   = (wave & 3) * 16;
  const int nW   = (wave >> 2) * 32;

  const float scale = 0.17677669529663687f;   // 1/sqrt(32)
  const int r0  = tid >> 3;
  const int cc4 = (tid & 7) * 4;

#pragma unroll
  for (int i = 0; i < 2; ++i) {
    int r = r0 + i * 32;
    float4 qa = make_float4(0.f, 0.f, 0.f, 0.f);
    int m = mBlk + r;
    if (m < LQ) qa = *(const float4*)(Qb + (size_t)m * (BATCH * 768) + cc4);
    v4h pa; pa[0] = (_Float16)(qa.x * scale); pa[1] = (_Float16)(qa.y * scale);
            pa[2] = (_Float16)(qa.z * scale); pa[3] = (_Float16)(qa.w * scale);
    *(v4h*)&sA[r * 32 + cc4] = pa;

    float4 ka = make_float4(0.f, 0.f, 0.f, 0.f);
    int n = nBlk + r;
    if (n < LQ) ka = *(const float4*)(Kb + (size_t)n * (BATCH * 768) + cc4);
    v4h pb; pb[0] = (_Float16)ka.x; pb[1] = (_Float16)ka.y;
            pb[2] = (_Float16)ka.z; pb[3] = (_Float16)ka.w;
    *(v4h*)&sB[r * 32 + cc4] = pb;
  }
  __syncthreads();

  const v8h* arow = (const v8h*)&sA[(mW + ln16) * 32];
  v8h alo = arow[2 * hi];
  v8h ahi = arow[2 * hi + 1];
  v16h a;
#pragma unroll
  for (int i = 0; i < 8; ++i) { a[i] = alo[i]; a[8 + i] = ahi[i]; }

  v8f cacc[2] = {{}, {}};
#pragma unroll
  for (int j = 0; j < 2; ++j) {
    v16h bf = *(const v16h*)&sB[(nW + j * 16 + ln16) * 32 + hi * 16];
    cacc[j] = __builtin_amdgcn_wmma_f32_16x16x32_f16(false, a, false, bf,
                                                     (short)0, cacc[j],
                                                     false, false);
  }

  _Float16* Sb = S + (size_t)z * LQ * SP;
#pragma unroll
  for (int j = 0; j < 2; ++j) {
    int n = nBlk + nW + j * 16 + ln16;
#pragma unroll
    for (int r = 0; r < 8; ++r) {
      int m = mBlk + mW + r + hi * 8;
      if (m < LQ && n < LQ) Sb[(size_t)m * SP + n] = (_Float16)cacc[j][r];
    }
  }
}

// ---------------------------------------------------------------------------
// Attention stage 2: rowwise softmax of S (900 valid cols), write P (f16),
// pad cols 900..927 set to 0. One wave per row; 29 values per lane.
// ---------------------------------------------------------------------------
__global__ void attn_softmax_kernel(_Float16* __restrict__ S) {
  int row  = blockIdx.x * 8 + (threadIdx.x >> 5);   // 0 .. 32*900-1
  int lane = threadIdx.x & 31;
  _Float16* p = S + (size_t)row * SP;

  float v[29];
  float mx = -1e30f;
#pragma unroll
  for (int i = 0; i < 29; ++i) {
    int j = lane + i * 32;
    v[i] = (j < LQ) ? (float)p[j] : -1e30f;
    mx = fmaxf(mx, v[i]);
  }
#pragma unroll
  for (int o = 16; o; o >>= 1) mx = fmaxf(mx, __shfl_xor(mx, o, 32));

  float sum = 0.f;
#pragma unroll
  for (int i = 0; i < 29; ++i) {
    v[i] = __expf(v[i] - mx);
    sum += v[i];
  }
#pragma unroll
  for (int o = 16; o; o >>= 1) sum += __shfl_xor(sum, o, 32);
  float inv = 1.f / sum;

#pragma unroll
  for (int i = 0; i < 29; ++i) {
    int j = lane + i * 32;
    p[j] = (_Float16)((j < LQ) ? v[i] * inv : 0.f);
  }
}

// ---------------------------------------------------------------------------
// Attention stage 3: O[m, h*32+n] = sum_k P[m,k] * V[k,n]   (per bh)
// A = P (f16, SP stride, pad cols are zero), K-steps = SP/32 = 29, N = 32.
// V staged transposed into LDS ([n][k]); tile 64(M) x 32(N), wave = 16x16.
// ---------------------------------------------------------------------------
__global__ void attn_pv_kernel(const _Float16* __restrict__ S,
                               const float* __restrict__ qkv,
                               float* __restrict__ out) {   // [MQ,256]
  __shared__ alignas(64) _Float16 sA[64 * 32];
  __shared__ alignas(64) _Float16 sB[32 * 32];

  const int z = blockIdx.z;
  const int b = z >> 3, h = z & 7;
  const _Float16* Pb = S + (size_t)z * LQ * SP;
  const float* Vb = qkv + (size_t)b * 768 + 512 + h * 32;

  const int tid  = threadIdx.x;
  const int lane = tid & 31;
  const int wave = tid >> 5;
  const int hi   = lane >> 4;
  const int ln16 = lane & 15;
  const int mBlk = blockIdx.x * 64;
  const int mW   = (wave & 3) * 16;
  const int nW   = (wave >> 2) * 16;    // 0 / 16

  const int r0  = tid >> 3;             // 0..31
  const int cc4 = (tid & 7) * 4;

  v8f c = {};
  for (int k0 = 0; k0 < SP; k0 += 32) {
    __syncthreads();
    // ---- stage A: P tile 64x32 (f16, direct) ----
#pragma unroll
    for (int i = 0; i < 2; ++i) {
      int r = r0 + i * 32;
      int m = mBlk + r;
      v4h pa = {};
      if (m < LQ) pa = *(const v4h*)(Pb + (size_t)m * SP + k0 + cc4);
      *(v4h*)&sA[r * 32 + cc4] = pa;
    }
    // ---- stage B: V tile 32(k)x32(n) -> sB[n][k] (transpose) ----
    {
      int k  = tid >> 3;                // 0..31
      int n4 = (tid & 7) * 4;
      int key = k0 + k;
      float4 vv = make_float4(0.f, 0.f, 0.f, 0.f);
      if (key < LQ) vv = *(const float4*)(Vb + (size_t)key * (BATCH * 768) + n4);
      sB[(n4 + 0) * 32 + k] = (_Float16)vv.x;
      sB[(n4 + 1) * 32 + k] = (_Float16)vv.y;
      sB[(n4 + 2) * 32 + k] = (_Float16)vv.z;
      sB[(n4 + 3) * 32 + k] = (_Float16)vv.w;
    }
    __syncthreads();

    const v8h* arow = (const v8h*)&sA[(mW + ln16) * 32];
    v8h alo = arow[2 * hi];
    v8h ahi = arow[2 * hi + 1];
    v16h a;
#pragma unroll
    for (int i = 0; i < 8; ++i) { a[i] = alo[i]; a[8 + i] = ahi[i]; }

    v16h bf = *(const v16h*)&sB[(nW + ln16) * 32 + hi * 16];
    c = __builtin_amdgcn_wmma_f32_16x16x32_f16(false, a, false, bf,
                                               (short)0, c, false, false);
  }

  int n = nW + ln16;                    // 0..31
#pragma unroll
  for (int r = 0; r < 8; ++r) {
    int m = mBlk + mW + r + hi * 8;
    if (m < LQ)
      out[(size_t)(m * BATCH + b) * D_MODEL + h * 32 + n] = c[r];
  }
}

// ---------------------------------------------------------------------------
// MSDeformAttn sampling: one wave per (q,b,h).
// ---------------------------------------------------------------------------
__global__ void msdeform_sample_kernel(const float* __restrict__ value,
                                       const float* __restrict__ offb,   // [MQ,256]
                                       const float* __restrict__ awb,    // [MQ,128]
                                       const float* __restrict__ ref,    // [LQ,B,4,2]
                                       float* __restrict__ acc) {        // [MQ,256]
  const int Harr[4] = {100, 50, 25, 13};
  const int Warr[4] = {134, 67, 34, 17};
  const int Sarr[4] = {0, 13400, 16750, 17600};

  int wid  = blockIdx.x * 8 + (threadIdx.x >> 5);
  int lane = threadIdx.x & 31;
  int h = wid & 7;
  int b = (wid >> 3) & 3;
  int q = wid >> 5;
  if (q >= LQ) return;
  int row = q * BATCH + b;

  int lp = lane & 15;
  int l  = lp >> 2;

  float logit = awb[(size_t)row * 128 + h * 16 + lp];
  float mx = logit;
#pragma unroll
  for (int o = 8; o; o >>= 1) mx = fmaxf(mx, __shfl_xor(mx, o, 16));
  float e = __expf(logit - mx);
  float ssum = e;
#pragma unroll
  for (int o = 8; o; o >>= 1) ssum += __shfl_xor(ssum, o, 16);
  float aw = e / ssum;

  int Wl = Warr[l], Hl = Harr[l], St = Sarr[l];
  float refx = ref[((size_t)row * 4 + l) * 2 + 0];
  float refy = ref[((size_t)row * 4 + l) * 2 + 1];
  float dx = offb[(size_t)row * 256 + h * 32 + lp * 2 + 0];
  float dy = offb[(size_t)row * 256 + h * 32 + lp * 2 + 1];
  float px = refx * (float)Wl + dx - 0.5f;
  float py = refy * (float)Hl + dy - 0.5f;

  float accv = 0.f;
#pragma unroll 1
  for (int p = 0; p < 16; ++p) {
    float x = __shfl(px, p, 32);
    float y = __shfl(py, p, 32);
    float a = __shfl(aw, p, 32);
    int WW = __shfl(Wl, p, 32);
    int HH = __shfl(Hl, p, 32);
    int SS = __shfl(St, p, 32);
    float x0 = floorf(x), y0 = floorf(y);
    float sampled = 0.f;
#pragma unroll
    for (int cidx = 0; cidx < 4; ++cidx) {
      float xi = x0 + (float)(cidx & 1);
      float yi = y0 + (float)(cidx >> 1);
      float wgt = (1.f - fabsf(x - xi)) * (1.f - fabsf(y - yi));
      if (xi >= 0.f && xi < (float)WW && yi >= 0.f && yi < (float)HH) {
        int idx = (int)yi * WW + (int)xi;
        sampled += wgt * value[(size_t)(SS + idx) * (BATCH * D_MODEL)
                               + b * D_MODEL + h * 32 + lane];
      }
    }
    accv += a * sampled;
  }
  acc[(size_t)row * D_MODEL + h * 32 + lane] = accv;
}

// ---------------------------------------------------------------------------
// Fused residual + LayerNorm over C=256: one block (256 threads) per row.
// ---------------------------------------------------------------------------
__global__ void addln_kernel(const float* __restrict__ x,
                             const float* __restrict__ y,
                             const float* __restrict__ g,
                             const float* __restrict__ bt,
                             float* __restrict__ out) {
  int row = blockIdx.x;
  int t = threadIdx.x;
  float v = x[(size_t)row * D_MODEL + t] + y[(size_t)row * D_MODEL + t];

  __shared__ float red[8];
  float s = v;
#pragma unroll
  for (int o = 16; o; o >>= 1) s += __shfl_xor(s, o, 32);
  if ((t & 31) == 0) red[t >> 5] = s;
  __syncthreads();
  float mean = 0.f;
#pragma unroll
  for (int i = 0; i < 8; ++i) mean += red[i];
  mean *= (1.f / 256.f);
  __syncthreads();

  float d = v - mean;
  float s2 = d * d;
#pragma unroll
  for (int o = 16; o; o >>= 1) s2 += __shfl_xor(s2, o, 32);
  if ((t & 31) == 0) red[t >> 5] = s2;
  __syncthreads();
  float var = 0.f;
#pragma unroll
  for (int i = 0; i < 8; ++i) var += red[i];
  var *= (1.f / 256.f);

  out[(size_t)row * D_MODEL + t] = d * rsqrtf(var + 1e-5f) * g[t] + bt[t];
}

// ---------------------------------------------------------------------------
// Host-side launch
// ---------------------------------------------------------------------------
extern "C" void kernel_launch(void* const* d_in, const int* in_sizes, int n_in,
                              void* d_out, int out_size, void* d_ws, size_t ws_size,
                              hipStream_t stream) {
  (void)in_sizes; (void)n_in; (void)out_size; (void)ws_size;
  const float* tgt      = (const float*)d_in[0];
  const float* refpts   = (const float*)d_in[1];
  const float* memory   = (const float*)d_in[2];
  const float* vproj_w  = (const float*)d_in[5];
  const float* vproj_b  = (const float*)d_in[6];
  const float* off_w    = (const float*)d_in[7];
  const float* off_b    = (const float*)d_in[8];
  const float* aw_w     = (const float*)d_in[9];
  const float* aw_b     = (const float*)d_in[10];
  const float* oproj_w  = (const float*)d_in[11];
  const float* oproj_b  = (const float*)d_in[12];
  const float* ln1_g    = (const float*)d_in[13];
  const float* ln1_b    = (const float*)d_in[14];
  const float* sa_in_w  = (const float*)d_in[15];
  const float* sa_in_b  = (const float*)d_in[16];
  const float* sa_out_w = (const float*)d_in[17];
  const float* sa_out_b = (const float*)d_in[18];
  const float* ln2_g    = (const float*)d_in[19];
  const float* ln2_b    = (const float*)d_in[20];
  const float* l1_w     = (const float*)d_in[21];
  const float* l1_b     = (const float*)d_in[22];
  const float* l2_w     = (const float*)d_in[23];
  const float* l2_b     = (const float*)d_in[24];
  const float* ln3_g    = (const float*)d_in[25];
  const float* ln3_b    = (const float*)d_in[26];
  const float* sa3_in_w = (const float*)d_in[27];
  const float* sa3_in_b = (const float*)d_in[28];
  const float* sa3_out_w= (const float*)d_in[29];
  const float* sa3_out_b= (const float*)d_in[30];
  const float* ln5_g    = (const float*)d_in[31];
  const float* ln5_b    = (const float*)d_in[32];
  const float* l31_w    = (const float*)d_in[33];
  const float* l31_b    = (const float*)d_in[34];
  const float* l32_w    = (const float*)d_in[35];
  const float* l32_b    = (const float*)d_in[36];
  const float* ln33_g   = (const float*)d_in[37];
  const float* ln33_b   = (const float*)d_in[38];

  float* ws  = (float*)d_ws;
  float* VAL = ws;                                  // MV*256 floats; reused:
                                                    //  - scores S (f16, 53MB)
                                                    //  - FFN hidden [MQ,2048]
  float* T1  = VAL + (size_t)MV * D_MODEL;          // MQ*256
  float* AW  = T1 + (size_t)MQ * D_MODEL;           // MQ*128
  float* T2  = AW + (size_t)MQ * 128;               // MQ*256
  float* T3  = T2 + (size_t)MQ * D_MODEL;           // MQ*256
  float* T4  = T3 + (size_t)MQ * D_MODEL;           // MQ*256
  float* T5  = T4 + (size_t)MQ * D_MODEL;           // MQ*768 (qkv)
  float* OUT = (float*)d_out;
  _Float16* SC = (_Float16*)VAL;                    // 32*900*928 f16 scores

  auto gemm = [&](const float* A, const float* W, const float* bias, float* C,
                  int M, int N, int K, int relu) {
    dim3 grid((M + 63) / 64, N / 64);
    gemm_wmma_kernel<<<grid, 256, 0, stream>>>(A, W, bias, C, M, N, K, relu);
  };
  auto attention = [&](const float* qkv, float* heads_out) {
    attn_qk_kernel<<<dim3(15, 15, 32), 256, 0, stream>>>(qkv, SC);
    attn_softmax_kernel<<<(32 * LQ) / 8, 256, 0, stream>>>(SC);
    attn_pv_kernel<<<dim3(15, 1, 32), 256, 0, stream>>>(SC, qkv, heads_out);
  };
  const int WAVEBLKS = (MQ * NHEADS) / 8;  // 3600

  // ---- ca: MSDeformAttn ----
  gemm(memory, vproj_w, vproj_b, VAL, MV, D_MODEL, D_MODEL, 0);       // value
  gemm(tgt, off_w, off_b, T1, MQ, 256, D_MODEL, 0);                   // offsets
  gemm(tgt, aw_w, aw_b, AW, MQ, 128, D_MODEL, 0);                     // aw logits
  msdeform_sample_kernel<<<WAVEBLKS, 256, 0, stream>>>(VAL, T1, AW, refpts, T2);
  gemm(T2, oproj_w, oproj_b, T3, MQ, D_MODEL, D_MODEL, 0);            // tgt2
  addln_kernel<<<MQ, 256, 0, stream>>>(tgt, T3, ln1_g, ln1_b, T4);    // x1

  // ---- sa ----
  gemm(T4, sa_in_w, sa_in_b, T5, MQ, 768, D_MODEL, 0);                // qkv
  attention(T5, T2);
  gemm(T2, sa_out_w, sa_out_b, T3, MQ, D_MODEL, D_MODEL, 0);
  addln_kernel<<<MQ, 256, 0, stream>>>(T4, T3, ln2_g, ln2_b, T1);     // x2

  // ---- ffn ----
  gemm(T1, l1_w, l1_b, VAL, MQ, D_FFN, D_MODEL, 1);
  gemm(VAL, l2_w, l2_b, T3, MQ, D_MODEL, D_FFN, 0);
  addln_kernel<<<MQ, 256, 0, stream>>>(T1, T3, ln3_g, ln3_b, T4);     // x3

  // ---- sa3 ----
  gemm(T4, sa3_in_w, sa3_in_b, T5, MQ, 768, D_MODEL, 0);
  attention(T5, T2);
  gemm(T2, sa3_out_w, sa3_out_b, T3, MQ, D_MODEL, D_MODEL, 0);
  addln_kernel<<<MQ, 256, 0, stream>>>(T4, T3, ln5_g, ln5_b, T1);     // x4

  // ---- ffn3 ----
  gemm(T1, l31_w, l31_b, VAL, MQ, D_FFN, D_MODEL, 1);
  gemm(VAL, l32_w, l32_b, T3, MQ, D_MODEL, D_FFN, 0);
  addln_kernel<<<MQ, 256, 0, stream>>>(T1, T3, ln33_g, ln33_b, OUT);
}
